// Radar_LSTM_17300128268949
// MI455X (gfx1250) — matrix operand order repaired and checked by reading, not compiled, not run on hardware
//
#include <hip/hip_runtime.h>
#include <hip/hip_bf16.h>

typedef _Float16 v16h __attribute__((ext_vector_type(16)));
typedef __fp16   h2_t __attribute__((ext_vector_type(2)));   // matches cvt_pkrtz return type
typedef float    v8f  __attribute__((ext_vector_type(8)));
typedef float    v4f  __attribute__((ext_vector_type(4)));

#define B_SZ  16384
#define T_SZ  512
#define I_SZ  3
#define H_SZ  15

// workspace byte offsets (prep kernel rewrites all of this every call)
#define WS_A_LSTM 0      // 4*32*16 f16 = 4096 B  (LSTM weights, A-layout images)
#define WS_A_W1   4096   // 4*32*16 f16           (head layer1 weights)
#define WS_A_W2   8192   // 2*4*32*16 f16 = 8192B (head layer2 weights, 2 K-chunks)
#define WS_B_LSTM 16384  // 4*32*8 f32            (LSTM bias, C/D-layout image)
#define WS_B_1    20480  // 4*32*8 f32
#define WS_B_2    24576  // 4*32*8 f32

// Map (lane, half-index) -> K column for the 16-bit A-matrix 16x32 layout:
// VGPR v: lanes 0-15 K=2v,2v+1 ; lanes 16-31 K=+8 ; VGPRs 4-7 are K+16.
__device__ __forceinline__ int k_of(int lane, int hv) {
  int v = hv >> 1, b = hv & 1, hi = lane >> 4;
  int k = (v < 4) ? (2 * v + b) : (16 + 2 * (v - 4) + b);
  return k + 8 * hi;
}

__global__ void prep_kernel(const float* __restrict__ W_ih, const float* __restrict__ W_hh,
                            const float* __restrict__ b_ih, const float* __restrict__ b_hh,
                            const float* __restrict__ W1,   const float* __restrict__ b1,
                            const float* __restrict__ W2,   const float* __restrict__ b2,
                            _Float16* __restrict__ wsA,  _Float16* __restrict__ wsA1,
                            _Float16* __restrict__ wsA2, float* __restrict__ wsBL,
                            float* __restrict__ wsB1,    float* __restrict__ wsB2) {
  const int tid = threadIdx.x;  // 256 threads
  if (tid < 128) {
    // LSTM weight image: K ordering  0..14 <- W_hh[:,0..14], 16..18 <- W_ih
    const int t = tid >> 5, lane = tid & 31, m = lane & 15, hi = lane >> 4;
    for (int hv = 0; hv < 16; ++hv) {
      int k = k_of(lane, hv);
      float w = 0.f;
      if (m < H_SZ) {
        int g = t * H_SZ + m;                       // torch gate row (i,f,g,o blocks of 15)
        if (k < H_SZ)               w = W_hh[g * H_SZ + k];
        else if (k >= 16 && k <= 18) w = W_ih[g * I_SZ + (k - 16)];
      }
      wsA[(t * 32 + lane) * 16 + hv] = (_Float16)w;
    }
    for (int r = 0; r < 8; ++r) {                   // bias images in C/D layout: M = r + 8*hi
      int mm = r + 8 * hi;
      wsBL[(t * 32 + lane) * 8 + r] = (mm < H_SZ) ? (b_ih[t * H_SZ + mm] + b_hh[t * H_SZ + mm]) : 0.f;
      wsB2[(t * 32 + lane) * 8 + r] = b2[t * 16 + mm];
    }
  } else {
    // Head layer-1 weight image: same h packing as LSTM (K 0..14 = h0..14)
    const int id = tid - 128;
    const int t = id >> 5, lane = id & 31, m = lane & 15, hi = lane >> 4;
    const int row = t * 16 + m;                     // D1 = 64, no padding
    for (int hv = 0; hv < 16; ++hv) {
      int k = k_of(lane, hv);
      float w = 0.f;
      if (k < H_SZ)               w = W1[row * H_SZ + k];
      wsA1[(t * 32 + lane) * 16 + hv] = (_Float16)w;
    }
    for (int r = 0; r < 8; ++r)
      wsB1[(t * 32 + lane) * 8 + r] = b1[t * 16 + r + 8 * hi];
  }
  // Head layer-2 weights, permuted so y1's D-layout feeds B with simple packs.
  {
    const int kc = tid >> 7, rem = tid & 127;
    const int t = rem >> 5, lane = rem & 31, m = lane & 15;
    const int row = t * 16 + m;
    for (int hv = 0; hv < 16; ++hv) {
      int kp = k_of(lane, hv);                     // 0..31 within chunk
      int src = kc * 32 + kp;
      wsA2[((kc * 4 + t) * 32 + lane) * 16 + hv] = (_Float16)W2[row * 64 + src];
    }
  }
}

__device__ __forceinline__ float tanh_f(float a) {
#if __has_builtin(__builtin_amdgcn_tanhf)
  return __builtin_amdgcn_tanhf(a);
#elif __has_builtin(__builtin_amdgcn_tanh_f32)
  return __builtin_amdgcn_tanh_f32(a);
#else
  float r;
  asm volatile("v_tanh_f32 %0, %1\n\tv_nop" : "=v"(r) : "v"(a));  // CDNA5 trans op + hazard slot
  return r;
#endif
}
__device__ __forceinline__ float sigm_f(float x) {
  return __builtin_fmaf(tanh_f(0.5f * x), 0.5f, 0.5f);
}

union Bu_t { v16h v; h2_t h[8]; };

__global__ __launch_bounds__(32)
void lstm_kernel(const float* __restrict__ x,
                 const _Float16* __restrict__ wsA,  const float* __restrict__ wsBL,
                 const _Float16* __restrict__ wsA1, const float* __restrict__ wsB1,
                 const _Float16* __restrict__ wsA2, const float* __restrict__ wsB2,
                 float* __restrict__ out) {
  const int lane  = threadIdx.x & 31;
  const int tile  = blockIdx.x;        // 1024 tiles of 16 batch rows, one wave each
  const int n     = lane & 15;
  const int hi    = lane >> 4;
  const int nBase = tile * 16;

  // Resident LSTM weights (A-layout, 4 gate tiles) and bias (C/D layout)
  v16h Aw0 = *(const v16h*)(wsA + (0 * 32 + lane) * 16);
  v16h Aw1 = *(const v16h*)(wsA + (1 * 32 + lane) * 16);
  v16h Aw2 = *(const v16h*)(wsA + (2 * 32 + lane) * 16);
  v16h Aw3 = *(const v16h*)(wsA + (3 * 32 + lane) * 16);
  v8f  C0  = *(const v8f*)(wsBL + (0 * 32 + lane) * 8);
  v8f  C1  = *(const v8f*)(wsBL + (1 * 32 + lane) * 8);
  v8f  C2  = *(const v8f*)(wsBL + (2 * 32 + lane) * 8);
  v8f  C3  = *(const v8f*)(wsBL + (3 * 32 + lane) * 8);

  v8f c = {};                      // cell state, D layout [16 hidden x 16 batch]
  v8f h = {};
  Bu_t B;
  h2_t z2 = (h2_t)0.0f;
#pragma unroll
  for (int v = 0; v < 8; ++v) B.h[v] = z2;   // h0 = 0, x slots filled below

  const float* px = x + (size_t)(nBase + n) * T_SZ * I_SZ;
  float x0 = px[0], x1 = px[1], x2 = px[2];

  for (int t = 0; t < T_SZ; ++t) {
    // x goes into K=16,17,18 (lanes 16-31 map to zeroed weight columns -> harmless)
    B.h[4] = __builtin_amdgcn_cvt_pkrtz(x0, x1);
    B.h[5] = __builtin_amdgcn_cvt_pkrtz(x2, 0.f);
    if (t + 1 < T_SZ) px += I_SZ;                 // prefetch next step's x
    float nx0 = px[0], nx1 = px[1], nx2 = px[2];

    v8f gi = __builtin_amdgcn_wmma_f32_16x16x32_f16(false, Aw0, false, B.v, (short)0, C0, false, false);
    v8f gf = __builtin_amdgcn_wmma_f32_16x16x32_f16(false, Aw1, false, B.v, (short)0, C1, false, false);
    v8f gg = __builtin_amdgcn_wmma_f32_16x16x32_f16(false, Aw2, false, B.v, (short)0, C2, false, false);
    v8f go = __builtin_amdgcn_wmma_f32_16x16x32_f16(false, Aw3, false, B.v, (short)0, C3, false, false);

#pragma unroll
    for (int r = 0; r < 8; ++r) {
      float si = sigm_f(gi[r]);
      float sf = sigm_f(gf[r]);
      float tg = tanh_f(gg[r]);
      float so = sigm_f(go[r]);
      float cr = __builtin_fmaf(sf, c[r], si * tg);
      c[r] = cr;
      h[r] = so * tanh_f(cr);
    }
    // D layout of h -> B layout of next step: pure in-lane packs (by construction)
#pragma unroll
    for (int v = 0; v < 4; ++v)
      B.h[v] = __builtin_amdgcn_cvt_pkrtz(h[2 * v], h[2 * v + 1]);
    x0 = nx0; x1 = nx1; x2 = nx2;
  }

  // ---- head: y1 = relu(W1 h + b1), out = W2 y1 + b2, all via WMMA ----
  B.h[4] = z2; B.h[5] = z2; B.h[6] = z2; B.h[7] = z2;
  v8f y[4];
#pragma unroll
  for (int tt = 0; tt < 4; ++tt) {
    v16h a  = *(const v16h*)(wsA1 + (tt * 32 + lane) * 16);
    v8f  bb = *(const v8f*)(wsB1 + (tt * 32 + lane) * 8);
    v8f  yy = __builtin_amdgcn_wmma_f32_16x16x32_f16(false, a, false, B.v, (short)0, bb, false, false);
#pragma unroll
    for (int r = 0; r < 8; ++r) yy[r] = fmaxf(yy[r], 0.f);
    y[tt] = yy;
  }
  Bu_t Bc0, Bc1;
#pragma unroll
  for (int v = 0; v < 4; ++v) {
    Bc0.h[v]     = __builtin_amdgcn_cvt_pkrtz(y[0][2 * v], y[0][2 * v + 1]);
    Bc0.h[v + 4] = __builtin_amdgcn_cvt_pkrtz(y[1][2 * v], y[1][2 * v + 1]);
    Bc1.h[v]     = __builtin_amdgcn_cvt_pkrtz(y[2][2 * v], y[2][2 * v + 1]);
    Bc1.h[v + 4] = __builtin_amdgcn_cvt_pkrtz(y[3][2 * v], y[3][2 * v + 1]);
  }
#pragma unroll
  for (int tt = 0; tt < 4; ++tt) {
    v16h a0  = *(const v16h*)(wsA2 + ((0 * 4 + tt) * 32 + lane) * 16);
    v16h a1  = *(const v16h*)(wsA2 + ((1 * 4 + tt) * 32 + lane) * 16);
    v8f  acc = *(const v8f*)(wsB2 + (tt * 32 + lane) * 8);
    acc = __builtin_amdgcn_wmma_f32_16x16x32_f16(false, a0, false, Bc0.v, (short)0, acc, false, false);
    acc = __builtin_amdgcn_wmma_f32_16x16x32_f16(false, a1, false, Bc1.v, (short)0, acc, false, false);
    // lane holds out[nBase+n, tt*16 + hi*8 + 0..7] -> two contiguous b128 stores
    float* po = out + (size_t)(nBase + n) * 64 + tt * 16 + hi * 8;
    v4f lo4; lo4[0] = acc[0]; lo4[1] = acc[1]; lo4[2] = acc[2]; lo4[3] = acc[3];
    v4f hi4; hi4[0] = acc[4]; hi4[1] = acc[5]; hi4[2] = acc[6]; hi4[3] = acc[7];
    *(v4f*)po       = lo4;
    *(v4f*)(po + 4) = hi4;
  }
}

extern "C" void kernel_launch(void* const* d_in, const int* in_sizes, int n_in,
                              void* d_out, int out_size, void* d_ws, size_t ws_size,
                              hipStream_t stream) {
  (void)in_sizes; (void)n_in; (void)out_size; (void)ws_size;
  const float* x    = (const float*)d_in[0];
  const float* W_ih = (const float*)d_in[1];
  const float* W_hh = (const float*)d_in[2];
  const float* b_ih = (const float*)d_in[3];
  const float* b_hh = (const float*)d_in[4];
  const float* W1   = (const float*)d_in[5];
  const float* b1   = (const float*)d_in[6];
  const float* W2   = (const float*)d_in[7];
  const float* b2   = (const float*)d_in[8];
  char* ws = (char*)d_ws;
  _Float16* wsA  = (_Float16*)(ws + WS_A_LSTM);
  _Float16* wsA1 = (_Float16*)(ws + WS_A_W1);
  _Float16* wsA2 = (_Float16*)(ws + WS_A_W2);
  float* wsBL = (float*)(ws + WS_B_LSTM);
  float* wsB1 = (float*)(ws + WS_B_1);
  float* wsB2 = (float*)(ws + WS_B_2);
  float* out  = (float*)d_out;

  prep_kernel<<<1, 256, 0, stream>>>(W_ih, W_hh, b_ih, b_hh, W1, b1, W2, b2,
                                     wsA, wsA1, wsA2, wsBL, wsB1, wsB2);
  lstm_kernel<<<B_SZ / 16, 32, 0, stream>>>(x, wsA, wsBL, wsA1, wsB1, wsA2, wsB2, out);
}
